// YourModel_26903675142282
// MI455X (gfx1250) — matrix operand order repaired and checked by reading, not compiled, not run on hardware
//
#include <hip/hip_runtime.h>
#include <hip/hip_bf16.h>
#include <math.h>

typedef __attribute__((ext_vector_type(16))) __bf16 v16bf;
typedef __attribute__((ext_vector_type(2)))  __bf16 v2bf;
typedef __attribute__((ext_vector_type(8)))  float  v8f;

#define B_TOT 4096
#define T_LEN 512
#define IN_D  11
#define H_D   32
#define G_D   128   // 4*H
#define TILE  16    // batch rows per workgroup
#define EPS   1e-5f

// LDS row paddings chosen so the 16 per-lane 16B regions of the A-fragment
// ds_load_b128 gathers are distinct modulo 256B (64 banks x 4B) -> conflict-free.
#define XP    24    // sX row: 24 bf16 = 48B stride
#define HP    40    // sH row: 40 bf16 = 80B stride
#define GP    (G_D + 2) // sG row: 130 floats = 520B stride

__device__ __forceinline__ __bf16 f2bf(float f) {
  union { float f; unsigned u; } v; v.f = f;
  unsigned r = (v.u + 0x7FFFu + ((v.u >> 16) & 1u)) >> 16;
  unsigned short s = (unsigned short)r;
  __bf16 o; __builtin_memcpy(&o, &s, 2); return o;
}

// Let the backend pick the native f32->bf16 FP_ROUND lowering (v_cvt_*bf16*).
__device__ __forceinline__ void store_bf16_pair(__bf16* p, float a, float b) {
  v2bf v;
  v[0] = (__bf16)a;
  v[1] = (__bf16)b;
  *(v2bf*)p = v;
}

// gfx1250 native v_tanh_f32 transcendental.
__device__ __forceinline__ float fast_tanh(float x) {
#if __has_builtin(__builtin_amdgcn_tanhf)
  return __builtin_amdgcn_tanhf(x);
#else
  float e = __expf(2.0f * x);
  return 1.0f - 2.0f * __builtin_amdgcn_rcpf(e + 1.0f);
#endif
}

__global__ __launch_bounds__(256, 1)
void lstm2_fused_wmma(const float* __restrict__ x,
                      const float* __restrict__ ln1_g, const float* __restrict__ ln1_b,
                      const float* __restrict__ Wih0, const float* __restrict__ Whh0,
                      const float* __restrict__ bih0, const float* __restrict__ bhh0,
                      const float* __restrict__ Wih1, const float* __restrict__ Whh1,
                      const float* __restrict__ bih1, const float* __restrict__ bhh1,
                      const float* __restrict__ ln2_g, const float* __restrict__ ln2_b,
                      const float* __restrict__ fc_w, const float* __restrict__ fc_b,
                      float* __restrict__ out)
{
  // Concatenated weights, B-matrix friendly [K=64][N=128] bf16.
  // Layer0: K 0..10 = Wih0 (IN=11), K 16..47 = Whh0, rest zero.
  // Layer1: K 0..31 = Wih1,        K 32..63 = Whh1.
  // Sigmoid-gate columns (n<64 or n>=96) are pre-scaled by 0.5 so the
  // activation is a bare tanh: sigmoid(x) = 0.5 + 0.5*tanh(0.5*x).
  __shared__ __bf16 sW0[64][G_D];
  __shared__ __bf16 sW1[64][G_D];
  __shared__ float  sBias0[G_D], sBias1[G_D];
  __shared__ __bf16 sX [TILE][XP];   // LN'd x_t tile, cols 0..15 used (11 + pad)
  __shared__ __bf16 sH0[TILE][HP];   // layer0 h (bf16, WMMA feed), cols 0..31 used
  __shared__ __bf16 sH1[TILE][HP];   // layer1 h
  __shared__ float  sC0[TILE][H_D], sC1[TILE][H_D];
  __shared__ float  sHf[TILE][H_D];  // f32 copy of layer1 h (for final LN)
  __shared__ float  sG [TILE][GP];   // activated gates
  __shared__ float  sLg[IN_D], sLb[IN_D];

  const int tid  = threadIdx.x;
  const int lane = tid & 31;
  const int wave = tid >> 5;          // 8 waves = 8 N-tiles of 16 gate columns
  const int b0   = blockIdx.x * TILE;

  // ---- preload weights / biases / LN params; zero state ----
  for (int idx = tid; idx < 64 * G_D; idx += 256) {
    int k = idx >> 7, n = idx & 127;
    float sc = (n >= 2 * H_D && n < 3 * H_D) ? 1.0f : 0.5f;  // g-gate cols keep scale 1
    float w0 = 0.f;
    if (k < IN_D)               w0 = Wih0[n * IN_D + k];
    else if (k >= 16 && k < 48) w0 = Whh0[n * H_D + (k - 16)];
    sW0[k][n] = f2bf(w0 * sc);
    float w1 = (k < 32) ? Wih1[n * H_D + k] : Whh1[n * H_D + (k - 32)];
    sW1[k][n] = f2bf(w1 * sc);
  }
  if (tid < G_D) {
    float sc = (tid >= 2 * H_D && tid < 3 * H_D) ? 1.0f : 0.5f;
    sBias0[tid] = (bih0[tid] + bhh0[tid]) * sc;
    sBias1[tid] = (bih1[tid] + bhh1[tid]) * sc;
  }
  if (tid < IN_D) { sLg[tid] = ln1_g[tid]; sLb[tid] = ln1_b[tid]; }
  for (int idx = tid; idx < TILE * H_D; idx += 256) {
    int m = idx >> 5, hc = idx & 31;
    sC0[m][hc] = 0.f; sC1[m][hc] = 0.f; sHf[m][hc] = 0.f;
    sH0[m][hc] = f2bf(0.f); sH1[m][hc] = f2bf(0.f);
  }
  if (tid < TILE)
    for (int k = IN_D; k < 16; ++k) sX[tid][k] = f2bf(0.f);
  __syncthreads();

  // ---- per-wave loop-invariant B fragments (32x16 bf16 each) ----
  // B layout: lanes 0-15: N=lane, K in {0..7,16..23} ; lanes 16-31: N=lane-16, K in {8..15,24..31}
  const int ncol = (wave << 4) + (lane & 15);
  const int klo  = (lane < 16) ? 0 : 8;
  v16bf b0lo, b0hi, b1lo, b1hi;
  #pragma unroll
  for (int j = 0; j < 16; ++j) {
    b0lo[j] = sW0[klo + j + (j & 8)][ncol];
    b0hi[j] = sW0[32 + klo + j + (j & 8)][ncol];
    b1lo[j] = sW1[klo + j + (j & 8)][ncol];
    b1hi[j] = sW1[32 + klo + j + (j & 8)][ncol];
  }
  // gate = act_off + act_amp * tanh(pre_scaled)
  //   tanh gate (waves 4,5): amp=1,   off=0
  //   sigmoid gates:         amp=0.5, off=0.5  (input pre-scaled by 0.5 in weights)
  const int   gateType = wave >> 1;       // 0=i 1=f 2=g(tanh) 3=o
  const float act_amp  = (gateType == 2) ? 1.0f : 0.5f;
  const float act_off  = (gateType == 2) ? 0.0f : 0.5f;
  const float bias0v   = sBias0[ncol];
  const float bias1v   = sBias1[ncol];
  // A layout: lanes 0-15: M=lane, K in {0..7,16..23}; lanes 16-31: K in {8..15,24..31}
  const int m    = lane & 15;
  const int akb  = (lane < 16) ? 0 : 8;
  const int moff = (lane < 16) ? 0 : 8;
  const __bf16 bz = f2bf(0.f);
  // state-update pair owned by this thread (adjacent columns, same row)
  const int umm = (tid * 2) >> 5;
  const int uhc = (tid * 2) & 31;

  for (int t = 0; t < T_LEN; ++t) {
    // ---- LayerNorm(x_t) -> sX (one lane per batch row) ----
    if (wave == 0 && lane < TILE) {
      const float* px = x + ((size_t)(b0 + lane) * T_LEN + t) * IN_D;
      float v[IN_D]; float mu = 0.f;
      #pragma unroll
      for (int k = 0; k < IN_D; ++k) { v[k] = px[k]; mu += v[k]; }
      mu *= (1.0f / IN_D);
      float var = 0.f;
      #pragma unroll
      for (int k = 0; k < IN_D; ++k) { float d = v[k] - mu; var += d * d; }
      float rs = rsqrtf(var * (1.0f / IN_D) + EPS);
      #pragma unroll
      for (int k = 0; k < IN_D; ++k)
        sX[lane][k] = f2bf((v[k] - mu) * rs * sLg[k] + sLb[k]);
      if (t + 1 < T_LEN) __builtin_prefetch(px + IN_D, 0, 1);
    }
    __syncthreads();

    // ---- layer 0 gates: [x_norm | h0_prev | 0] (K=64) x W0cat ----
    {
      v16bf alo, ahi;
      #pragma unroll
      for (int j = 0; j < 8; ++j) {
        alo[j]     = sX [m][akb + j];        // K 0..15
        alo[8 + j] = sH0[m][akb + j];        // K 16..31
        ahi[j]     = sH0[m][16 + akb + j];   // K 32..47
        ahi[8 + j] = bz;                     // K 48..63 pad
      }
      v8f c;
      #pragma unroll
      for (int r = 0; r < 8; ++r) c[r] = bias0v;
      c = __builtin_amdgcn_wmma_f32_16x16x32_bf16(false, alo, false, b0lo, (short)0, c, false, false);
      c = __builtin_amdgcn_wmma_f32_16x16x32_bf16(false, ahi, false, b0hi, (short)0, c, false, false);
      #pragma unroll
      for (int r = 0; r < 8; ++r)
        sG[r + moff][ncol] = fmaf(act_amp, fast_tanh(c[r]), act_off);
    }
    __syncthreads();

    // ---- layer 0 state update (2 adjacent elems per thread) ----
    {
      float ig0 = sG[umm][uhc],           ig1 = sG[umm][uhc + 1];
      float fg0 = sG[umm][H_D + uhc],     fg1 = sG[umm][H_D + uhc + 1];
      float gg0 = sG[umm][2 * H_D + uhc], gg1 = sG[umm][2 * H_D + uhc + 1];
      float og0 = sG[umm][3 * H_D + uhc], og1 = sG[umm][3 * H_D + uhc + 1];
      float cn0 = fg0 * sC0[umm][uhc]     + ig0 * gg0;
      float cn1 = fg1 * sC0[umm][uhc + 1] + ig1 * gg1;
      sC0[umm][uhc] = cn0; sC0[umm][uhc + 1] = cn1;
      store_bf16_pair(&sH0[umm][uhc], og0 * fast_tanh(cn0), og1 * fast_tanh(cn1));
    }
    __syncthreads();

    // ---- layer 1 gates: [h0_t | h1_prev] (K=64) x W1cat ----
    {
      v16bf alo, ahi;
      #pragma unroll
      for (int j = 0; j < 8; ++j) {
        alo[j]     = sH0[m][akb + j];
        alo[8 + j] = sH0[m][16 + akb + j];
        ahi[j]     = sH1[m][akb + j];
        ahi[8 + j] = sH1[m][16 + akb + j];
      }
      v8f c;
      #pragma unroll
      for (int r = 0; r < 8; ++r) c[r] = bias1v;
      c = __builtin_amdgcn_wmma_f32_16x16x32_bf16(false, alo, false, b1lo, (short)0, c, false, false);
      c = __builtin_amdgcn_wmma_f32_16x16x32_bf16(false, ahi, false, b1hi, (short)0, c, false, false);
      #pragma unroll
      for (int r = 0; r < 8; ++r)
        sG[r + moff][ncol] = fmaf(act_amp, fast_tanh(c[r]), act_off);
    }
    __syncthreads();

    // ---- layer 1 state update ----
    {
      float ig0 = sG[umm][uhc],           ig1 = sG[umm][uhc + 1];
      float fg0 = sG[umm][H_D + uhc],     fg1 = sG[umm][H_D + uhc + 1];
      float gg0 = sG[umm][2 * H_D + uhc], gg1 = sG[umm][2 * H_D + uhc + 1];
      float og0 = sG[umm][3 * H_D + uhc], og1 = sG[umm][3 * H_D + uhc + 1];
      float cn0 = fg0 * sC1[umm][uhc]     + ig0 * gg0;
      float cn1 = fg1 * sC1[umm][uhc + 1] + ig1 * gg1;
      float hn0 = og0 * fast_tanh(cn0);
      float hn1 = og1 * fast_tanh(cn1);
      sC1[umm][uhc] = cn0; sC1[umm][uhc + 1] = cn1;
      sHf[umm][uhc] = hn0; sHf[umm][uhc + 1] = hn1;
      store_bf16_pair(&sH1[umm][uhc], hn0, hn1);
    }
    __syncthreads();
  }

  // ---- LayerNorm(h1_last) -> ReLU -> FC (one lane per row) ----
  if (wave == 0 && lane < TILE) {
    float mu = 0.f;
    #pragma unroll
    for (int k = 0; k < H_D; ++k) mu += sHf[lane][k];
    mu *= (1.0f / H_D);
    float var = 0.f;
    #pragma unroll
    for (int k = 0; k < H_D; ++k) { float d = sHf[lane][k] - mu; var += d * d; }
    float rs = rsqrtf(var * (1.0f / H_D) + EPS);
    float acc = fc_b[0];
    #pragma unroll
    for (int k = 0; k < H_D; ++k) {
      float yv = (sHf[lane][k] - mu) * rs * ln2_g[k] + ln2_b[k];
      acc += fmaxf(yv, 0.f) * fc_w[k];
    }
    out[b0 + lane] = acc;
  }
}

extern "C" void kernel_launch(void* const* d_in, const int* in_sizes, int n_in,
                              void* d_out, int out_size, void* d_ws, size_t ws_size,
                              hipStream_t stream) {
  (void)in_sizes; (void)n_in; (void)out_size; (void)d_ws; (void)ws_size;
  const float* x     = (const float*)d_in[0];
  const float* ln1_g = (const float*)d_in[1];
  const float* ln1_b = (const float*)d_in[2];
  const float* Wih0  = (const float*)d_in[3];
  const float* Whh0  = (const float*)d_in[4];
  const float* bih0  = (const float*)d_in[5];
  const float* bhh0  = (const float*)d_in[6];
  const float* Wih1  = (const float*)d_in[7];
  const float* Whh1  = (const float*)d_in[8];
  const float* bih1  = (const float*)d_in[9];
  const float* bhh1  = (const float*)d_in[10];
  const float* ln2_g = (const float*)d_in[11];
  const float* ln2_b = (const float*)d_in[12];
  const float* fc_w  = (const float*)d_in[13];
  const float* fc_b  = (const float*)d_in[14];
  float* out = (float*)d_out;

  dim3 grid(B_TOT / TILE);   // 256 workgroups
  dim3 block(256);           // 8 wave32 waves = 8 gate N-tiles
  lstm2_fused_wmma<<<grid, block, 0, stream>>>(
      x, ln1_g, ln1_b, Wih0, Whh0, bih0, bhh0,
      Wih1, Whh1, bih1, bhh1, ln2_g, ln2_b, fc_w, fc_b, out);
}
